// AttentionHeads_31490700214825
// MI455X (gfx1250) — matrix operand order repaired and checked by reading, not compiled, not run on hardware
//
#include <hip/hip_runtime.h>

typedef __attribute__((ext_vector_type(16))) __bf16 v16bf;
typedef __attribute__((ext_vector_type(8)))  float  v8f;
typedef __attribute__((ext_vector_type(4)))  int    v4i;

// ---------------------------------------------------------------------------
// Async global->LDS copy (CDNA5 GLOBAL_LOAD_ASYNC_TO_LDS_B128, ASYNCcnt),
// guarded so absence of the builtin falls back to a plain 16B copy.
// Probe result: arg0 is int4* in AS(1) (global), arg1 LDS side, then imm, imm.
// ---------------------------------------------------------------------------
#if __has_builtin(__builtin_amdgcn_global_load_async_to_lds_b128)
#define ASYNC_G2L 1
#endif

__device__ inline void cp16(void* lds, const void* gl) {
#ifdef ASYNC_G2L
  __builtin_amdgcn_global_load_async_to_lds_b128(
      (__attribute__((address_space(1))) v4i*)(void*)gl,
      (__attribute__((address_space(3))) v4i*)lds, 0, 0);
#else
  *(float4*)lds = *(const float4*)gl;
#endif
}

__device__ inline void async_join() {
#ifdef ASYNC_G2L
#if __has_builtin(__builtin_amdgcn_s_wait_asynccnt)
  __builtin_amdgcn_s_wait_asynccnt(0);
#else
  asm volatile("s_wait_asynccnt 0" ::: "memory");
#endif
#endif
}

// ---------------------------------------------------------------------------
// WMMA fragment loaders (V_WMMA_F32_16X16X32_BF16, wave32), ISA 7.12.2 layouts
// ---------------------------------------------------------------------------
__device__ inline v16bf load_a_frag(const __bf16* base, int stride, int lane) {
  int m    = lane & 15;
  int koff = (lane >> 4) << 3;   // 0 or 8
  const __bf16* p = base + m * stride;
  v16bf a;
#pragma unroll
  for (int i = 0; i < 8; ++i) {
    a[i]     = p[koff + i];
    a[8 + i] = p[16 + koff + i];
  }
  return a;
}

__device__ inline v16bf load_b_frag(const __bf16* baseT, int stride, int lane) {
  int n    = lane & 15;
  int koff = (lane >> 4) << 3;   // 0 or 8
  const __bf16* p = baseT + n * stride;
  v16bf b;
#pragma unroll
  for (int i = 0; i < 8; ++i) {
    b[i]     = p[koff + i];
    b[8 + i] = p[16 + koff + i];
  }
  return b;
}

// ---------------------------------------------------------------------------
// One-time precision/layout conversion kernels (tiny vs. 140 GFLOP of GEMM)
// ---------------------------------------------------------------------------
__global__ __launch_bounds__(256)
void cvt_f32_to_bf16(const float* __restrict__ in, __bf16* __restrict__ out, size_t n) {
  size_t i = (size_t)blockIdx.x * blockDim.x + threadIdx.x;
  size_t stride = (size_t)gridDim.x * blockDim.x;
  for (; i < n; i += stride) out[i] = (__bf16)in[i];
}

// W[K][N] f32 -> WT[N][K] bf16
__global__ __launch_bounds__(256)
void transpose_cvt(const float* __restrict__ W, __bf16* __restrict__ WT, int K, int N) {
  __shared__ float tile[32][33];
  int bx = blockIdx.x * 32;          // N
  int by = blockIdx.y * 32;          // K
  int tx = threadIdx.x & 31;
  int ty = threadIdx.x >> 5;         // 0..7
#pragma unroll
  for (int r = ty; r < 32; r += 8)
    tile[r][tx] = W[(size_t)(by + r) * N + bx + tx];
  __syncthreads();
#pragma unroll
  for (int r = ty; r < 32; r += 8)
    WT[(size_t)(bx + r) * K + by + tx] = (__bf16)tile[tx][r];
}

// ---------------------------------------------------------------------------
// GEMM + bias, bf16 operands, f32 accumulate:
//   Y[M,N] = A[M,K] @ BT[N,K]^T + bias,  Y is bf16 or f32.
// Block tile 256x128, K-step 64, 256 threads = 8 waves (4x2), wave tile 64x64
// -> 32 WMMA per wave per barrier, 1:1 ds_load_b128 : wmma.
// ---------------------------------------------------------------------------
#define GT_M 256
#define GT_N 128
#define GT_K 64
#define GPAD 8

template <bool OUT_BF16>
__global__ __launch_bounds__(256)
void gemm_bf16(const __bf16* __restrict__ A,    // [M][K]
               const __bf16* __restrict__ BT,   // [N][K]
               const float*  __restrict__ bias,
               void* __restrict__ Yv, int M, int N, int K) {
  __shared__ __bf16 As[GT_M][GT_K + GPAD];      // 256 x 72 -> 36 KB
  __shared__ __bf16 Bs[GT_N][GT_K + GPAD];      // 128 x 72 -> 18 KB

  const int tid  = threadIdx.x;
  const int lane = tid & 31;
  const int wave = tid >> 5;
  const int wm   = wave >> 1;   // 0..3 -> 64-row slab
  const int wn   = wave & 1;    // 0..1 -> 64-col slab
  const int m0   = blockIdx.y * GT_M;
  const int n0   = blockIdx.x * GT_N;

  v8f acc[4][4] = {};

  for (int k0 = 0; k0 < K; k0 += GT_K) {
    {   // stage A: 256 rows x 64 bf16 (128B/row), one row per thread
      const __bf16* src = A + (size_t)(m0 + tid) * K + k0;
#pragma unroll
      for (int c = 0; c < 8; ++c) cp16(&As[tid][c * 8], src + c * 8);
    }
    {   // stage BT: 128 rows x 64 bf16, two threads per row
      int n  = tid >> 1;
      int c0 = (tid & 1) * 32;
      const __bf16* src = BT + (size_t)(n0 + n) * K + k0 + c0;
#pragma unroll
      for (int c = 0; c < 4; ++c) cp16(&Bs[n][c0 + c * 8], src + c * 8);
    }
    async_join();
    __syncthreads();

#pragma unroll
    for (int kc = 0; kc < GT_K; kc += 32) {
      v16bf bfr[4];
#pragma unroll
      for (int nt = 0; nt < 4; ++nt)
        bfr[nt] = load_b_frag(&Bs[wn * 64 + nt * 16][kc], GT_K + GPAD, lane);
#pragma unroll
      for (int mt = 0; mt < 4; ++mt) {
        v16bf a = load_a_frag(&As[wm * 64 + mt * 16][kc], GT_K + GPAD, lane);
#pragma unroll
        for (int nt = 0; nt < 4; ++nt) {
          acc[mt][nt] = __builtin_amdgcn_wmma_f32_16x16x32_bf16(
              false, a, false, bfr[nt], (short)0, acc[mt][nt], false, false);
        }
      }
    }
    __syncthreads();
  }

  const int nlane = lane & 15;
  const int rbase = (lane >> 4) * 8;
#pragma unroll
  for (int mt = 0; mt < 4; ++mt) {
#pragma unroll
    for (int nt = 0; nt < 4; ++nt) {
      int col  = n0 + wn * 64 + nt * 16 + nlane;
      float bv = bias[col];
#pragma unroll
      for (int r = 0; r < 8; ++r) {
        int row = m0 + wm * 64 + mt * 16 + rbase + r;
        float v = acc[mt][nt][r] + bv;
        if (OUT_BF16) ((__bf16*)Yv)[(size_t)row * N + col] = (__bf16)v;
        else          ((float*)Yv)[(size_t)row * N + col]  = v;
      }
    }
  }
}

// ---------------------------------------------------------------------------
// Flash attention (single shared K/V head). bf16 Q/K/V in, bf16 attn-out.
// grid (S/64, HQ, B), 128 threads = 4 waves, 64-key tiles, f32 online softmax.
// ---------------------------------------------------------------------------
#define AQ 64
#define AK 64
#define DHD 128
#define APAD 8

__global__ __launch_bounds__(128)
void flash_attn_bf16wmma(const __bf16* __restrict__ Q, const __bf16* __restrict__ Kb,
                         const __bf16* __restrict__ Vb, __bf16* __restrict__ O,
                         int S, int D) {
  __shared__ __bf16 Qs[AQ][DHD + APAD];
  __shared__ __bf16 Ks[AK][DHD + APAD];
  __shared__ __bf16 VsT[DHD][AK + APAD];
  __shared__ __bf16 Ps[4][16][AK + APAD];

  const int tid  = threadIdx.x;
  const int lane = tid & 31;
  const int wave = tid >> 5;
  const int q0   = blockIdx.x * AQ;
  const int h    = blockIdx.y;
  const int b    = blockIdx.z;

  const __bf16* Qp = Q  + (size_t)b * S * D + (size_t)h * DHD;
  const __bf16* Kp = Kb + (size_t)b * S * DHD;
  const __bf16* Vp = Vb + (size_t)b * S * DHD;

  {   // stage Q tile 64 x 128 bf16: two threads per row, 8x16B each
    int row = tid >> 1;
    int c0  = (tid & 1) * 64;
    const __bf16* src = Qp + (size_t)(q0 + row) * D + c0;
#pragma unroll
    for (int c = 0; c < 8; ++c) cp16(&Qs[row][c0 + c * 8], src + c * 8);
  }

  const float scale = 0.08838834764831845f;   // 1/sqrt(128)
  const int nlane = lane & 15;
  const int rb    = (lane >> 4) * 8;

  v8f oacc[8] = {};
  float mrow[8], lrow[8];
#pragma unroll
  for (int r = 0; r < 8; ++r) { mrow[r] = -1e30f; lrow[r] = 0.f; }

  for (int kt = 0; kt < S; kt += AK) {
    {   // stage K (copy) and V (transposed, manual) tiles
      int row = tid >> 1;
      int c0  = (tid & 1) * 64;
      const __bf16* ksrc = Kp + (size_t)(kt + row) * DHD + c0;
      const __bf16* vsrc = Vp + (size_t)(kt + row) * DHD + c0;
#pragma unroll
      for (int c = 0; c < 8; ++c) cp16(&Ks[row][c0 + c * 8], ksrc + c * 8);
#pragma unroll
      for (int i = 0; i < 64; ++i) VsT[c0 + i][row] = vsrc[i];
    }
    async_join();
    __syncthreads();

    // S = Q K^T : wave computes 16x64 (4 n-tiles), DH=128 -> 4 k-chunks
    v8f sacc[4] = {};
#pragma unroll
    for (int kc = 0; kc < DHD; kc += 32) {
      v16bf a = load_a_frag(&Qs[wave * 16][kc], DHD + APAD, lane);
#pragma unroll
      for (int nt = 0; nt < 4; ++nt) {
        v16bf bfr = load_b_frag(&Ks[nt * 16][kc], DHD + APAD, lane);
        sacc[nt] = __builtin_amdgcn_wmma_f32_16x16x32_bf16(
            false, a, false, bfr, (short)0, sacc[nt], false, false);
      }
    }
#pragma unroll
    for (int nt = 0; nt < 4; ++nt)
#pragma unroll
      for (int r = 0; r < 8; ++r) sacc[nt][r] *= scale;

    // online softmax (rows live across the 16 lanes of a half-wave)
    float alpha[8];
#pragma unroll
    for (int r = 0; r < 8; ++r) {
      float mx = sacc[0][r];
#pragma unroll
      for (int nt = 1; nt < 4; ++nt) mx = fmaxf(mx, sacc[nt][r]);
#pragma unroll
      for (int sh = 1; sh < 16; sh <<= 1) mx = fmaxf(mx, __shfl_xor(mx, sh, 32));
      float mnew = fmaxf(mrow[r], mx);
      alpha[r] = __expf(mrow[r] - mnew);
      mrow[r]  = mnew;
    }

    float psum[8];
#pragma unroll
    for (int r = 0; r < 8; ++r) psum[r] = 0.f;
#pragma unroll
    for (int nt = 0; nt < 4; ++nt) {
#pragma unroll
      for (int r = 0; r < 8; ++r) {
        float p = __expf(sacc[nt][r] - mrow[r]);
        psum[r] += p;
        Ps[wave][rb + r][nt * 16 + nlane] = (__bf16)p;   // wave-local
      }
    }
#pragma unroll
    for (int r = 0; r < 8; ++r) {
      float s2 = psum[r];
#pragma unroll
      for (int sh = 1; sh < 16; sh <<= 1) s2 += __shfl_xor(s2, sh, 32);
      lrow[r] = lrow[r] * alpha[r] + s2;
    }
#pragma unroll
    for (int t = 0; t < 8; ++t)
#pragma unroll
      for (int r = 0; r < 8; ++r) oacc[t][r] *= alpha[r];

    // O += P @ V : 8 dh-tiles, key dim 64 -> 2 k-chunks
#pragma unroll
    for (int kc = 0; kc < AK; kc += 32) {
      v16bf a = load_a_frag(&Ps[wave][0][kc], AK + APAD, lane);
#pragma unroll
      for (int nt = 0; nt < 8; ++nt) {
        v16bf bfr = load_b_frag(&VsT[nt * 16][kc], AK + APAD, lane);
        oacc[nt] = __builtin_amdgcn_wmma_f32_16x16x32_bf16(
            false, a, false, bfr, (short)0, oacc[nt], false, false);
      }
    }
    __syncthreads();
  }

  __bf16* Op = O + (size_t)b * S * D + (size_t)h * DHD;
#pragma unroll
  for (int r = 0; r < 8; ++r) lrow[r] = 1.0f / lrow[r];
#pragma unroll
  for (int nt = 0; nt < 8; ++nt) {
    int col = nt * 16 + nlane;
#pragma unroll
    for (int r = 0; r < 8; ++r) {
      int row = q0 + wave * 16 + rb + r;
      Op[(size_t)row * D + col] = (__bf16)(oacc[nt][r] * lrow[r]);
    }
  }
}

// ---------------------------------------------------------------------------
extern "C" void kernel_launch(void* const* d_in, const int* in_sizes, int n_in,
                              void* d_out, int out_size, void* d_ws, size_t ws_size,
                              hipStream_t stream) {
  const float* x  = (const float*)d_in[0];
  const float* Wq = (const float*)d_in[1];
  const float* bq = (const float*)d_in[2];
  const float* Wk = (const float*)d_in[3];
  const float* bk = (const float*)d_in[4];
  const float* Wv = (const float*)d_in[5];
  const float* bv = (const float*)d_in[6];
  const float* Wo = (const float*)d_in[7];
  const float* bo = (const float*)d_in[8];
  float* out = (float*)d_out;

  const int B = 2, S = 2048, D = 2048, DH = 128, HQ = 16;
  const int M = B * S;

  // bf16 workspace layout (elements)
  __bf16* xb  = (__bf16*)d_ws;                    // M*D
  __bf16* WqT = xb  + (size_t)M * D;              // D*D    [N][K]
  __bf16* WkT = WqT + (size_t)D * D;              // DH*D
  __bf16* WvT = WkT + (size_t)DH * D;             // DH*D
  __bf16* WoT = WvT + (size_t)DH * D;             // D*D
  __bf16* Qb  = WoT + (size_t)D * D;              // M*D
  __bf16* Kbf = Qb  + (size_t)M * D;              // M*DH
  __bf16* Vbf = Kbf + (size_t)M * DH;             // M*DH
  __bf16* Ob  = Vbf + (size_t)M * DH;             // M*D

  // one-time conversions
  cvt_f32_to_bf16<<<dim3(2048), dim3(256), 0, stream>>>(x, xb, (size_t)M * D);
  transpose_cvt<<<dim3(D / 32,  D / 32), dim3(256), 0, stream>>>(Wq, WqT, D, D);
  transpose_cvt<<<dim3(DH / 32, D / 32), dim3(256), 0, stream>>>(Wk, WkT, D, DH);
  transpose_cvt<<<dim3(DH / 32, D / 32), dim3(256), 0, stream>>>(Wv, WvT, D, DH);
  transpose_cvt<<<dim3(D / 32,  D / 32), dim3(256), 0, stream>>>(Wo, WoT, D, D);

  dim3 blk(256);
  // projections (bf16 out)
  gemm_bf16<true><<<dim3(D / GT_N,  M / GT_M), blk, 0, stream>>>(xb, WqT, bq, Qb,  M, D,  D);
  gemm_bf16<true><<<dim3(DH / GT_N, M / GT_M), blk, 0, stream>>>(xb, WkT, bk, Kbf, M, DH, D);
  gemm_bf16<true><<<dim3(DH / GT_N, M / GT_M), blk, 0, stream>>>(xb, WvT, bv, Vbf, M, DH, D);
  // flash attention
  flash_attn_bf16wmma<<<dim3(S / AQ, HQ, B), dim3(128), 0, stream>>>(Qb, Kbf, Vbf, Ob, S, D);
  // output projection (f32 out)
  gemm_bf16<false><<<dim3(D / GT_N, M / GT_M), blk, 0, stream>>>(Ob, WoT, bo, out, M, D, D);
}
